// EfficientGlobalPointer_10977936409146
// MI455X (gfx1250) — matrix operand order repaired and checked by reading, not compiled, not run on hardware
//
#include <hip/hip_runtime.h>
#include <hip/hip_bf16.h>

// CDNA5 / gfx1250: wave32, WMMA 16x16x32 bf16 path + LDS staging.
typedef __attribute__((ext_vector_type(16))) __bf16 v16bf;
typedef __attribute__((ext_vector_type(8)))  __bf16 v8bf;
typedef __attribute__((ext_vector_type(4)))  __bf16 v4bf;
typedef __attribute__((ext_vector_type(8)))  float  v8f;

#define BS     8
#define SEQ    512
#define HID    768
#define HS     64
#define NHEAD  12
#define XN     128            // W1 output columns (head_size*2)
#define MROWS  (BS*SEQ)       // 4096
#define INF_   10000.0f

// ---------------------------------------------------------------------------
// Repack W1 (768x128 f32, row-major) into bf16 B-fragment order:
// W1b[((kk*2 + half)*128 + n)*16 + j] = bf16(W1[(kk*32 + half*16 + j)*128 + n])
// so each lane's B fragment is 16 contiguous bf16 (32B) -> global_load_b128 x2.
// ---------------------------------------------------------------------------
__global__ void pack_w1(const float* __restrict__ W1, __bf16* __restrict__ W1b) {
    int idx = blockIdx.x * blockDim.x + threadIdx.x;       // HID*XN = 98304
    if (idx >= HID * XN) return;
    int j  = idx & 15;
    int n  = (idx >> 4) & 127;
    int h  = (idx >> 11) & 1;
    int kk = idx >> 12;
    W1b[idx] = (__bf16)W1[(size_t)(kk * 32 + h * 16 + j) * XN + n];
}

// ---------------------------------------------------------------------------
// X = inputs @ W1 + b1  via v_wmma_f32_16x16x32_bf16.
// Grid: 256 blocks (one 16-row M tile each), 128 threads = 4 waves.
// The block's 16x768 fp32 A tile is staged ONCE into LDS as bf16 (24 KB of
// the WGP's 320 KB), then every wave builds A fragments with ds_load_b128.
// Each wave owns two 16x16 N tiles; A is hoisted out of the N loop by keeping
// two accumulators, so each kk chunk does one A load feeding two WMMAs.
// ---------------------------------------------------------------------------
__global__ void __launch_bounds__(128) gemm1(const float* __restrict__ inp,
                                             const __bf16* __restrict__ W1b,
                                             const float* __restrict__ b1,
                                             float* __restrict__ X) {
    __shared__ __bf16 atile[16 * HID];            // 24 KB
    const int tid   = threadIdx.x;
    const int lane  = tid & 31;
    const int wave  = tid >> 5;
    const int mtile = blockIdx.x;                 // 0..255
    const int half  = lane >> 4;
    const int lidx  = lane & 15;

    // Cooperative global->LDS stage with fp32->bf16 convert (once per block).
    const float* gsrc = inp + (size_t)mtile * 16 * HID;
    for (int e = tid * 4; e < 16 * HID; e += 128 * 4) {
        const float4 v = *(const float4*)(gsrc + e);
        v4bf p;
        p[0] = (__bf16)v.x; p[1] = (__bf16)v.y;
        p[2] = (__bf16)v.z; p[3] = (__bf16)v.w;
        *(v4bf*)(atile + e) = p;                  // 8B-aligned ds_store
    }
    __syncthreads();

    const __bf16* arow = atile + (size_t)lidx * HID;   // this lane's A row
    const int nt0 = wave, nt1 = wave + 4;
    v8f acc0 = {}, acc1 = {};
    for (int kk = 0; kk < 24; ++kk) {
        const int base = kk * 32 + half * 16;
        // A fragment: lanes 0-15 hold K {0..15}, 16-31 hold {16..31}
        v8bf lo = *(const v8bf*)(arow + base);          // ds_load_b128
        v8bf hi = *(const v8bf*)(arow + base + 8);
        v16bf a;
#pragma unroll
        for (int i = 0; i < 8; ++i) { a[i] = lo[i]; a[8 + i] = hi[i]; }

        const __bf16* bp = W1b + (size_t)(kk * 2 + half) * 128 * 16;
        v16bf bf0 = *(const v16bf*)(bp + (size_t)(nt0 * 16 + lidx) * 16);
        v16bf bf1 = *(const v16bf*)(bp + (size_t)(nt1 * 16 + lidx) * 16);
        acc0 = __builtin_amdgcn_wmma_f32_16x16x32_bf16(
            false, a, false, bf0, (short)0, acc0, false, false);
        acc1 = __builtin_amdgcn_wmma_f32_16x16x32_bf16(
            false, a, false, bf1, (short)0, acc1, false, false);
    }

#pragma unroll
    for (int t = 0; t < 2; ++t) {
        const int   nt = t ? nt1 : nt0;
        const v8f&  ac = t ? acc1 : acc0;
        const int   n  = nt * 16 + lidx;
        const float bb = b1[n];
#pragma unroll
        for (int r = 0; r < 8; ++r) {             // D row r -> M = r + 8*half
            const int mrow = mtile * 16 + r + half * 8;
            X[(size_t)mrow * XN + n] = ac[r] + bb;
        }
    }
}

// ---------------------------------------------------------------------------
// RoPE + deinterleave even/odd columns of X into q,k (bf16, 4096x64 row-major).
// ---------------------------------------------------------------------------
__global__ void rope_pack(const float* __restrict__ X,
                          __bf16* __restrict__ q, __bf16* __restrict__ k) {
    int idx = blockIdx.x * blockDim.x + threadIdx.x;       // MROWS*32
    if (idx >= MROWS * 32) return;
    const int m   = idx >> 5;
    const int i   = idx & 31;                               // rotary pair index
    const int pos = m & (SEQ - 1);
    // inv_freq = 10000^(-2i/64) = exp(-(i/32)*ln(10000))
    const float inv = __expf(-(float)i * (9.210340371976184f / 32.0f));
    const float f   = (float)pos * inv;
    float s, c;
    __sincosf(f, &s, &c);
    const float* xr = X + (size_t)m * XN + 4 * i;
    const float q1 = xr[0], k1 = xr[1], q2 = xr[2], k2 = xr[3];
    __bf16* qo = q + (size_t)m * HS + 2 * i;
    __bf16* ko = k + (size_t)m * HS + 2 * i;
    qo[0] = (__bf16)(q1 * c - q2 * s);
    qo[1] = (__bf16)(q1 * s + q2 * c);
    ko[0] = (__bf16)(k1 * c - k2 * s);
    ko[1] = (__bf16)(k1 * s + k2 * c);
}

// ---------------------------------------------------------------------------
// bias[b, c, pos] = (X[b*512+pos, :] . W2[:, c] + b2[c]) * 0.5    (8,24,512)
// ---------------------------------------------------------------------------
__global__ void bias_kernel(const float* __restrict__ X,
                            const float* __restrict__ W2,
                            const float* __restrict__ b2,
                            float* __restrict__ bias) {
    int idx = blockIdx.x * blockDim.x + threadIdx.x;       // MROWS*24
    if (idx >= MROWS * 24) return;
    const int c = idx % 24;
    const int g = idx / 24;
    const float* xr = X + (size_t)g * XN;
    float acc = b2[c];
    for (int kk = 0; kk < XN; ++kk)
        acc += xr[kk] * W2[kk * 24 + c];
    const int b = g >> 9, pos = g & (SEQ - 1);
    bias[((size_t)b * 24 + c) * SEQ + pos] = acc * 0.5f;
}

// ---------------------------------------------------------------------------
// logits[b,h,m,n] = (q[m].k[n])/8 + bias[b,2h,n] + bias[b,2h+1,m]
//                   - (pad_mask + causal)*INF ; mask_out written once.
// Grid: 256 blocks = (b, mtile); 4 waves; each wave owns 8 ntiles and reuses
// its q A-fragments across them. Two bf16 WMMAs per tile (K=64).
// ---------------------------------------------------------------------------
__global__ void __launch_bounds__(128) attn(const __bf16* __restrict__ q,
                                            const __bf16* __restrict__ k,
                                            const float* __restrict__ bias,
                                            const int* __restrict__ am,
                                            float* __restrict__ out) {
    const int lane  = threadIdx.x & 31;
    const int wave  = threadIdx.x >> 5;
    const int b     = blockIdx.x >> 5;
    const int mtile = blockIdx.x & 31;
    const int half  = lane >> 4;
    const int lidx  = lane & 15;

    // --- A fragments (q row, K=64 split into two 32-chunks) ---
    const int mrow = b * SEQ + mtile * 16 + lidx;
    const __bf16* qrow = q + (size_t)mrow * HS;
    v16bf a0, a1;
    {
        v8bf c0 = *(const v8bf*)(qrow + half * 16);
        v8bf c1 = *(const v8bf*)(qrow + half * 16 + 8);
        v8bf c2 = *(const v8bf*)(qrow + 32 + half * 16);
        v8bf c3 = *(const v8bf*)(qrow + 32 + half * 16 + 8);
#pragma unroll
        for (int i = 0; i < 8; ++i) {
            a0[i] = c0[i]; a0[8 + i] = c1[i];
            a1[i] = c2[i]; a1[8 + i] = c3[i];
        }
    }
    // Row indices + padding mask for the 8 accumulator rows this lane owns.
    int   mg[8];
    float amm[8];
#pragma unroll
    for (int r = 0; r < 8; ++r) {
        mg[r]  = mtile * 16 + r + half * 8;
        amm[r] = (float)am[b * SEQ + mg[r]];
    }
    float* mask_out = out + (size_t)BS * NHEAD * SEQ * SEQ;
    const float* be_base = bias + (size_t)b * 24 * SEQ;

    for (int nt = wave; nt < 32; nt += 4) {       // uniform per wave
        const int ng = nt * 16 + lidx;
        const __bf16* krow = k + (size_t)(b * SEQ + ng) * HS;
        // prefetch the next n-tile's k row (speculative, OOB-safe)
        __builtin_prefetch(krow + 4 * 16 * HS, 0, 1);
        // B fragments: lanes 0-15 take K rows 0..15 of chunk, 16-31 take 16..31
        v16bf b0 = *(const v16bf*)(krow + half * 16);
        v16bf b1 = *(const v16bf*)(krow + 32 + half * 16);
        v8f acc = {};
        acc = __builtin_amdgcn_wmma_f32_16x16x32_bf16(
            false, a0, false, b0, (short)0, acc, false, false);
        acc = __builtin_amdgcn_wmma_f32_16x16x32_bf16(
            false, a1, false, b1, (short)0, acc, false, false);

        const float amn = (float)am[b * SEQ + ng];
        float lg[8];
#pragma unroll
        for (int r = 0; r < 8; ++r) {
            const float maskv = 1.0f - amn * amm[r];
            const float tril  = (ng < mg[r]) ? 1.0f : 0.0f;
            lg[r] = acc[r] * 0.125f - (maskv + tril) * INF_;
            mask_out[((size_t)b * SEQ + mg[r]) * SEQ + ng] =
                (maskv != 0.0f || tril != 0.0f) ? 1.0f : 0.0f;
        }
        // Broadcast the tile across 12 heads with per-head bias adds.
        for (int h = 0; h < NHEAD; ++h) {
            const float  be = be_base[(2 * h) * SEQ + ng];
            const float* bo = be_base + (2 * h + 1) * SEQ;
            float* orow = out + ((size_t)(b * NHEAD + h) * SEQ) * SEQ + ng;
#pragma unroll
            for (int r = 0; r < 8; ++r)
                orow[(size_t)mg[r] * SEQ] = lg[r] + be + bo[mg[r]];
        }
    }
}

// ---------------------------------------------------------------------------
extern "C" void kernel_launch(void* const* d_in, const int* in_sizes, int n_in,
                              void* d_out, int out_size, void* d_ws, size_t ws_size,
                              hipStream_t stream) {
    (void)in_sizes; (void)n_in; (void)out_size; (void)ws_size;
    const float* inputs = (const float*)d_in[0];
    const int*   amask  = (const int*)d_in[1];
    const float* W1     = (const float*)d_in[2];
    const float* b1     = (const float*)d_in[3];
    const float* W2     = (const float*)d_in[4];
    const float* b2     = (const float*)d_in[5];
    float* out = (float*)d_out;

    // Workspace layout (all offsets 256B-aligned by construction): ~3.7 MB
    char*  ws  = (char*)d_ws;
    size_t off = 0;
    float*  X    = (float*)(ws + off);  off += (size_t)MROWS * XN * sizeof(float);
    __bf16* W1b  = (__bf16*)(ws + off); off += (size_t)HID * XN * sizeof(__bf16);
    __bf16* qb   = (__bf16*)(ws + off); off += (size_t)MROWS * HS * sizeof(__bf16);
    __bf16* kb   = (__bf16*)(ws + off); off += (size_t)MROWS * HS * sizeof(__bf16);
    float*  bias = (float*)(ws + off);  off += (size_t)BS * 24 * SEQ * sizeof(float);

    pack_w1<<<(HID * XN + 255) / 256, 256, 0, stream>>>(W1, W1b);
    gemm1<<<256, 128, 0, stream>>>(inputs, W1b, b1, X);
    rope_pack<<<(MROWS * 32 + 255) / 256, 256, 0, stream>>>(X, qb, kb);
    bias_kernel<<<(MROWS * 24 + 255) / 256, 256, 0, stream>>>(X, W2, b2, bias);
    attn<<<256, 128, 0, stream>>>(qb, kb, bias, amask, out);
}
